// Multi_Modal_Aggregator_15857019257044
// MI455X (gfx1250) — matrix operand order repaired and checked by reading, not compiled, not run on hardware
//
#include <hip/hip_runtime.h>
#include <math.h>

// ---------------------------------------------------------------------------
// Types for CDNA5 WMMA (wave32): V_WMMA_F32_16X16X32_BF16
// ---------------------------------------------------------------------------
typedef __attribute__((ext_vector_type(16))) __bf16 v16bf;
typedef __attribute__((ext_vector_type(8)))  float  v8f;

// ---------------------------------------------------------------------------
// Tiled bf16 GEMM, fp32 accumulate, double-buffered LDS pipeline (2x unrolled
// so ping/pong LDS bases are compile-time constants).
//   C[M,N] = act( scale * (A @ opB) + bias )
//   A: [M,K] row-major bf16 (lda)
//   TRANS_B=1: B is [N,K] row-major (C = A @ B^T)   -> staged transposed
//   TRANS_B=0: B is [K,N] row-major (C = A @ B)     -> staged direct
// Block: 128x128 tile, 256 threads = 8 waves; wave tile 32x64 = 2x4 WMMA.
// One barrier per K-step: stage tile k+1 into the other buffer while
// computing tile k.
// ---------------------------------------------------------------------------
constexpr int BM = 128;
constexpr int BN = 128;
constexpr int BK = 32;
constexpr int LDS_A = BK + 8;   // 40 elems -> 80B row stride (16B aligned)
constexpr int LDS_B = BN + 8;   // 136 elems -> 272B row stride (16B aligned)

template<int ACT>
__device__ __forceinline__ float activate(float x) {
  if constexpr (ACT == 1) return x * (1.0f / (1.0f + __expf(-x)));  // SiLU
  else if constexpr (ACT == 2) return 1.0f / (1.0f + __expf(-x));   // sigmoid
  else return x;
}

union FragU { v16bf v; uint4 u[2]; };

template<bool TRANS_B, int ACT>
__global__ __launch_bounds__(256)
void gemm_bf16_wmma(const __bf16* __restrict__ A, int lda,
                    const __bf16* __restrict__ B, int ldb,
                    float* __restrict__ C, int ldc,
                    const float* __restrict__ bias,
                    float scale, int M, int N, int K)
{
  __shared__ __align__(16) __bf16 sA[2][BM * LDS_A];
  __shared__ __align__(16) __bf16 sB[2][BK * LDS_B];   // [k][n]

  const int tid  = threadIdx.x;
  const int lane = tid & 31;
  const int wave = tid >> 5;     // 0..7
  const int wm   = wave >> 1;    // 0..3 : 32-row slab
  const int wn   = wave & 1;     // 0..1 : 64-col slab

  const int bm = blockIdx.y * BM;
  const int bn = blockIdx.x * BN;

  const v8f vzero = {0.f,0.f,0.f,0.f,0.f,0.f,0.f,0.f};
  v8f acc[2][4];
#pragma unroll
  for (int mt = 0; mt < 2; ++mt)
#pragma unroll
    for (int nt = 0; nt < 4; ++nt) acc[mt][nt] = vzero;

  // -------- stage one K-slab (global -> LDS) into buffer sAb/sBb --------
  auto stage = [&](int kb, __bf16* sAb, __bf16* sBb) {
    // A tile: 128 rows x 32 cols
    {
      const int row  = tid >> 1;
      const int half = tid & 1;
      const int gr = bm + row;
      const int gc = kb + half * 16;
      __bf16* dst = &sAb[row * LDS_A + half * 16];
      if (gr < M && gc + 16 <= K) {
        const uint4* src = (const uint4*)(A + (size_t)gr * lda + gc);
        uint4 d0 = src[0];
        uint4 d1 = src[1];
        *(uint4*)(dst)     = d0;
        *(uint4*)(dst + 8) = d1;
      } else {
#pragma unroll
        for (int j = 0; j < 16; ++j) {
          float v = 0.f;
          if (gr < M && gc + j < K) v = (float)A[(size_t)gr * lda + gc + j];
          dst[j] = (__bf16)v;
        }
      }
    }
    // B tile into sBb[k][n]
    if constexpr (TRANS_B) {
      // B is [N,K]: read 16 contiguous K per thread, scatter into columns
      const int n    = tid >> 1;
      const int half = tid & 1;
      const int gn = bn + n;
      const int gk = kb + half * 16;
      __bf16 tmp[16];
      if (gn < N && gk + 16 <= K) {
        union { uint4 u[2]; __bf16 h[16]; } ld;
        const uint4* src = (const uint4*)(B + (size_t)gn * ldb + gk);
        ld.u[0] = src[0];
        ld.u[1] = src[1];
#pragma unroll
        for (int j = 0; j < 16; ++j) tmp[j] = ld.h[j];
      } else {
#pragma unroll
        for (int j = 0; j < 16; ++j) {
          float v = 0.f;
          if (gn < N && gk + j < K) v = (float)B[(size_t)gn * ldb + gk + j];
          tmp[j] = (__bf16)v;
        }
      }
#pragma unroll
      for (int j = 0; j < 16; ++j) sBb[(half * 16 + j) * LDS_B + n] = tmp[j];
    } else {
      // B is [K,N]: contiguous rows, direct copy
      const int k   = tid >> 3;          // 0..31
      const int seg = tid & 7;           // 8 x 16 cols
      const int gk = kb + k;
      const int gn = bn + seg * 16;
      __bf16* dst = &sBb[k * LDS_B + seg * 16];
      if (gk < K && gn + 16 <= N) {
        const uint4* src = (const uint4*)(B + (size_t)gk * ldb + gn);
        uint4 d0 = src[0];
        uint4 d1 = src[1];
        *(uint4*)(dst)     = d0;
        *(uint4*)(dst + 8) = d1;
      } else {
#pragma unroll
        for (int j = 0; j < 16; ++j) {
          float v = 0.f;
          if (gk < K && gn + j < N) v = (float)B[(size_t)gk * ldb + gn + j];
          dst[j] = (__bf16)v;
        }
      }
    }
  };

  // -------- compute one staged K-slab: fragments per ISA VGPR layout --------
  // A 16x32 bf16: lane<16 -> M=lane, K {0..7,16..23}; lane>=16 -> K {8..15,24..31}
  // B 32x16 bf16: lane<16 -> N=lane, K {0..7,16..23}; lane>=16 -> K {8..15,24..31}
  auto compute = [&](const __bf16* sAb, const __bf16* sBb) {
    v16bf afrag[2];
    {
      const int m   = lane & 15;
      const int kb0 = (lane >> 4) << 3;   // 0 or 8
#pragma unroll
      for (int mt = 0; mt < 2; ++mt) {
        const int r = wm * 32 + mt * 16 + m;
        FragU f;
        f.u[0] = *(const uint4*)(&sAb[r * LDS_A + kb0]);
        f.u[1] = *(const uint4*)(&sAb[r * LDS_A + kb0 + 16]);
        afrag[mt] = f.v;
      }
    }
    v16bf bfrag[4];
#pragma unroll
    for (int nt = 0; nt < 4; ++nt) {
      const int n   = wn * 64 + nt * 16 + (lane & 15);
      const int kb0 = (lane >> 4) << 3;
      v16bf b;
#pragma unroll
      for (int j = 0; j < 8; ++j) {
        b[j]     = sBb[(kb0 + j) * LDS_B + n];
        b[8 + j] = sBb[(kb0 + 16 + j) * LDS_B + n];
      }
      bfrag[nt] = b;
    }
#pragma unroll
    for (int mt = 0; mt < 2; ++mt)
#pragma unroll
      for (int nt = 0; nt < 4; ++nt)
        acc[mt][nt] = __builtin_amdgcn_wmma_f32_16x16x32_bf16(
            false, afrag[mt], false, bfrag[nt],
            (short)0, acc[mt][nt], false, false);
  };

  auto l2_prefetch = [&](int kslab) {
    if (kslab < K) {
      int gr = bm + (tid >> 1);
      if (gr < M) __builtin_prefetch(A + (size_t)gr * lda + kslab, 0, 1);
    }
  };

  // -------- software pipeline, 2x unrolled so LDS bases are constant --------
  stage(0, sA[0], sB[0]);
  __syncthreads();

  int kb = 0;
  for (;;) {
    // compute buffer 0, stage kb+BK into buffer 1
    {
      const int kn = kb + BK;
      if (kn < K) {
        l2_prefetch(kn + BK);
        stage(kn, sA[1], sB[1]);
      }
      compute(sA[0], sB[0]);
      __syncthreads();
      kb = kn;
      if (kb >= K) break;
    }
    // compute buffer 1, stage kb+BK into buffer 0
    {
      const int kn = kb + BK;
      if (kn < K) {
        l2_prefetch(kn + BK);
        stage(kn, sA[0], sB[0]);
      }
      compute(sA[1], sB[1]);
      __syncthreads();
      kb = kn;
      if (kb >= K) break;
    }
  }

  // -------- epilogue: C VGPR i -> row i + 8*(lane/16), col lane%16 --------
#pragma unroll
  for (int mt = 0; mt < 2; ++mt)
#pragma unroll
    for (int nt = 0; nt < 4; ++nt) {
      const int c = bn + wn * 64 + nt * 16 + (lane & 15);
      if (c >= N) continue;
      const float bv = bias ? bias[c] : 0.0f;
      const int rbase = bm + wm * 32 + mt * 16 + ((lane >> 4) << 3);
#pragma unroll
      for (int i = 0; i < 8; ++i) {
        const int r = rbase + i;
        if (r < M) {
          float x = acc[mt][nt][i] * scale + bv;
          C[(size_t)r * ldc + c] = activate<ACT>(x);
        }
      }
    }
}

// ---------------------------------------------------------------------------
// Row softmax over [4096, 4096]; one block per row; row cached in registers.
// ---------------------------------------------------------------------------
template<typename OUT>
__global__ __launch_bounds__(256)
void row_softmax_4096(const float* __restrict__ in, OUT* __restrict__ out)
{
  __shared__ float red[256];
  const int row = blockIdx.x;
  const int tid = threadIdx.x;
  const float* p = in + (size_t)row * 4096;
  float v[16];
  float mx = -3.4e38f;
#pragma unroll
  for (int i = 0; i < 16; ++i) { v[i] = p[tid + i * 256]; mx = fmaxf(mx, v[i]); }
  red[tid] = mx; __syncthreads();
  for (int s = 128; s > 0; s >>= 1) {
    if (tid < s) red[tid] = fmaxf(red[tid], red[tid + s]);
    __syncthreads();
  }
  mx = red[0]; __syncthreads();
  float sum = 0.f;
#pragma unroll
  for (int i = 0; i < 16; ++i) { v[i] = __expf(v[i] - mx); sum += v[i]; }
  red[tid] = sum; __syncthreads();
  for (int s = 128; s > 0; s >>= 1) {
    if (tid < s) red[tid] += red[tid + s];
    __syncthreads();
  }
  const float inv = 1.0f / red[0];
  OUT* q = out + (size_t)row * 4096;
#pragma unroll
  for (int i = 0; i < 16; ++i) q[tid + i * 256] = (OUT)(v[i] * inv);
}

// ---------------------------------------------------------------------------
// Elementwise glue kernels
// ---------------------------------------------------------------------------
__global__ void k_f32_to_bf16(const float* __restrict__ x, __bf16* __restrict__ y, int n) {
  int i = blockIdx.x * 256 + threadIdx.x;
  if (i < n) y[i] = (__bf16)x[i];
}

// attn = relu(sim)^2 * mask -> bf16
__global__ void k_attn(const float* __restrict__ sim, const float* __restrict__ mask,
                       __bf16* __restrict__ out, int n) {
  int i = blockIdx.x * 256 + threadIdx.x;
  if (i < n) {
    float r = fmaxf(sim[i], 0.0f);
    out[i] = (__bf16)(r * r * mask[i]);
  }
}

// q = qk*gamma0+beta0 ; k = qk*gamma1+beta1 (bf16 out), width 200
__global__ void k_qk_split(const float* __restrict__ qk,
                           const float* __restrict__ gamma, const float* __restrict__ beta,
                           __bf16* __restrict__ qo, __bf16* __restrict__ ko, int n) {
  int i = blockIdx.x * 256 + threadIdx.x;
  if (i < n) {
    int c = i % 200;
    float x = qk[i];
    qo[i] = (__bf16)(x * gamma[c]       + beta[c]);
    ko[i] = (__bf16)(x * gamma[200 + c] + beta[200 + c]);
  }
}

// out = (attn@v) * gate, gate = h[:, 768:1536]
__global__ void k_avgate(const float* __restrict__ av, const float* __restrict__ h,
                         __bf16* __restrict__ out, int n) {
  int i = blockIdx.x * 256 + threadIdx.x;
  if (i < n) {
    int r = i / 768, c = i % 768;
    out[i] = (__bf16)(av[i] * h[(size_t)r * 1536 + 768 + c]);
  }
}

// y = bf16(a * b)
__global__ void k_mul_bf16(const float* __restrict__ a, const float* __restrict__ b,
                           __bf16* __restrict__ y, int n) {
  int i = blockIdx.x * 256 + threadIdx.x;
  if (i < n) y[i] = (__bf16)(a[i] * b[i]);
}

// B = K * (Q + relu(G)) * Q   (fp32 + bf16 outputs)
__global__ void k_bcombine(const float* __restrict__ Kf, const float* __restrict__ Qf,
                           const float* __restrict__ Gf,
                           float* __restrict__ Bf, __bf16* __restrict__ Bbf, int n) {
  int i = blockIdx.x * 256 + threadIdx.x;
  if (i < n) {
    float t = Kf[i] * (Qf[i] + fmaxf(Gf[i], 0.0f)) * Qf[i];
    Bf[i] = t;
    Bbf[i] = (__bf16)t;
  }
}

// X1 = gt*K ; X2 = (1-gt)*Q   (bf16 outputs)
__global__ void k_gtsplit(const float* __restrict__ gt, const float* __restrict__ Kf,
                          const float* __restrict__ Qf,
                          __bf16* __restrict__ X1, __bf16* __restrict__ X2, int n) {
  int i = blockIdx.x * 256 + threadIdx.x;
  if (i < n) {
    float g = gt[i];
    X1[i] = (__bf16)(g * Kf[i]);
    X2[i] = (__bf16)((1.0f - g) * Qf[i]);
  }
}

// out = normalize(sigmoid(B1*B1*G)), per row of 256
__global__ __launch_bounds__(256)
void k_finalize(const float* __restrict__ B1, const float* __restrict__ G,
                float* __restrict__ out) {
  __shared__ float red[256];
  const int row = blockIdx.x;
  const int c = threadIdx.x;
  const size_t idx = (size_t)row * 256 + c;
  const float b = B1[idx];
  const float v = 1.0f / (1.0f + __expf(-(b * b * G[idx])));
  red[c] = v * v; __syncthreads();
  for (int s = 128; s > 0; s >>= 1) {
    if (c < s) red[c] += red[c + s];
    __syncthreads();
  }
  const float nrm = sqrtf(red[0]);
  out[idx] = v / fmaxf(nrm, 1e-12f);
}

// ---------------------------------------------------------------------------
// Orchestration
// ---------------------------------------------------------------------------
#define GEMM(TB, ACTV, Ap, lda, Bp, ldb, Cp, ldc, biasp, sc, Mv, Nv, Kv)            \
  gemm_bf16_wmma<TB, ACTV><<<dim3(((Nv) + BN - 1) / BN, ((Mv) + BM - 1) / BM),      \
                             dim3(256), 0, stream>>>(                               \
      Ap, lda, Bp, ldb, Cp, ldc, biasp, sc, Mv, Nv, Kv)

extern "C" void kernel_launch(void* const* d_in, const int* in_sizes, int n_in,
                              void* d_out, int out_size, void* d_ws, size_t ws_size,
                              hipStream_t stream) {
  (void)in_sizes; (void)n_in; (void)out_size; (void)ws_size;

  const int Nn = 4096, DIN = 256, DSUP = 768, H2 = 1536, QKD = 200;

  const float* node   = (const float*)d_in[0];
  const float* sup    = (const float*)d_in[1];
  const float* Wq     = (const float*)d_in[2];
  const float* bq     = (const float*)d_in[3];
  const float* Wk     = (const float*)d_in[4];
  const float* bk     = (const float*)d_in[5];
  const float* Wv     = (const float*)d_in[6];
  const float* bv     = (const float*)d_in[7];
  const float* Wg     = (const float*)d_in[8];
  const float* bg     = (const float*)d_in[9];
  const float* gWh    = (const float*)d_in[10];
  const float* gbh    = (const float*)d_in[11];
  const float* gWqk   = (const float*)d_in[12];
  const float* gbqk   = (const float*)d_in[13];
  const float* ggamma = (const float*)d_in[14];
  const float* gbeta  = (const float*)d_in[15];
  const float* gWout  = (const float*)d_in[16];
  const float* gbout  = (const float*)d_in[17];

  // Bump allocator over workspace
  char* w = (char*)d_ws;
  auto alloc = [&](size_t bytes) -> char* {
    char* p = w;
    w += (bytes + 255) & ~(size_t)255;
    return p;
  };

  __bf16* sup_bf  = (__bf16*)alloc((size_t)Nn * DSUP * 2);
  __bf16* node_bf = (__bf16*)alloc((size_t)Nn * DIN * 2);
  __bf16* Wq_bf   = (__bf16*)alloc((size_t)DIN * DIN * 2);
  __bf16* Wk_bf   = (__bf16*)alloc((size_t)DIN * DSUP * 2);
  __bf16* Wv_bf   = (__bf16*)alloc((size_t)DIN * DSUP * 2);
  __bf16* Wg_bf   = (__bf16*)alloc((size_t)DIN * DIN * 2);
  __bf16* Wh_bf   = (__bf16*)alloc((size_t)H2 * DSUP * 2);
  __bf16* Wqk_bf  = (__bf16*)alloc((size_t)QKD * DSUP * 2);
  __bf16* Wout_bf = (__bf16*)alloc((size_t)DSUP * DSUP * 2);

  float*  NN0   = (float*)alloc((size_t)Nn * Nn * 4);
  float*  NN1   = (float*)alloc((size_t)Nn * Nn * 4);
  __bf16* NNbf  = (__bf16*)alloc((size_t)Nn * Nn * 2);

  float*  h_f   = (float*)alloc((size_t)Nn * H2 * 4);
  __bf16* h_bf  = (__bf16*)alloc((size_t)Nn * H2 * 2);
  float*  qk_f  = (float*)alloc((size_t)Nn * QKD * 4);
  __bf16* q_bf  = (__bf16*)alloc((size_t)Nn * QKD * 2);
  __bf16* k_bf  = (__bf16*)alloc((size_t)Nn * QKD * 2);
  float*  av_f  = (float*)alloc((size_t)Nn * DSUP * 4);
  __bf16* avg_bf= (__bf16*)alloc((size_t)Nn * DSUP * 2);
  float*  o_f   = (float*)alloc((size_t)Nn * DSUP * 4);
  __bf16* s_bf  = (__bf16*)alloc((size_t)Nn * DSUP * 2);

  float*  K_f   = (float*)alloc((size_t)Nn * DIN * 4);
  float*  V_f   = (float*)alloc((size_t)Nn * DIN * 4);
  float*  Q_f   = (float*)alloc((size_t)Nn * DIN * 4);
  __bf16* K_bf  = (__bf16*)alloc((size_t)Nn * DIN * 2);
  __bf16* V_bf  = (__bf16*)alloc((size_t)Nn * DIN * 2);
  __bf16* Q_bf  = (__bf16*)alloc((size_t)Nn * DIN * 2);
  float*  G_f   = (float*)alloc((size_t)Nn * DIN * 4);
  float*  B1_f  = (float*)alloc((size_t)Nn * DIN * 4);
  float*  B2_f  = (float*)alloc((size_t)Nn * DIN * 4);
  __bf16* B1_bf = (__bf16*)alloc((size_t)Nn * DIN * 2);
  __bf16* B2_bf = (__bf16*)alloc((size_t)Nn * DIN * 2);
  float*  gt_f  = (float*)alloc((size_t)Nn * DIN * 4);
  __bf16* X1_bf = (__bf16*)alloc((size_t)Nn * DIN * 2);
  __bf16* X2_bf = (__bf16*)alloc((size_t)Nn * DIN * 2);
  float*  GB2_f = (float*)alloc((size_t)Nn * DIN * 4);

  auto cvt = [&](const float* src, __bf16* dst, int n) {
    k_f32_to_bf16<<<dim3((n + 255) / 256), dim3(256), 0, stream>>>(src, dst, n);
  };

  // --- bf16 copies of GEMM operands ---
  cvt(sup, sup_bf, Nn * DSUP);
  cvt(node, node_bf, Nn * DIN);
  cvt(Wq, Wq_bf, DIN * DIN);
  cvt(Wk, Wk_bf, DIN * DSUP);
  cvt(Wv, Wv_bf, DIN * DSUP);
  cvt(Wg, Wg_bf, DIN * DIN);
  cvt(gWh, Wh_bf, H2 * DSUP);
  cvt(gWqk, Wqk_bf, QKD * DSUP);
  cvt(gWout, Wout_bf, DSUP * DSUP);

  // --- mask = softmax(support @ support^T) ---
  GEMM(true, 0, sup_bf, DSUP, sup_bf, DSUP, NN0, Nn, (const float*)nullptr, 1.0f, Nn, Nn, DSUP);
  row_softmax_4096<float><<<Nn, 256, 0, stream>>>(NN0, NN0);

  // --- GAU: h = silu(x@Wh^T+bh); qk = silu(x@Wqk^T+bqk) ---
  GEMM(true, 1, sup_bf, DSUP, Wh_bf, DSUP, h_f, H2, gbh, 1.0f, Nn, H2, DSUP);
  GEMM(true, 1, sup_bf, DSUP, Wqk_bf, DSUP, qk_f, QKD, gbqk, 1.0f, Nn, QKD, DSUP);
  k_qk_split<<<dim3((Nn * QKD + 255) / 256), dim3(256), 0, stream>>>(
      qk_f, ggamma, gbeta, q_bf, k_bf, Nn * QKD);

  // --- sim = (q@k^T)/sqrt(768); attn = relu(sim)^2 * mask ---
  GEMM(true, 0, q_bf, QKD, k_bf, QKD, NN1, Nn, (const float*)nullptr,
       1.0f / sqrtf(768.0f), Nn, Nn, QKD);
  k_attn<<<dim3((Nn * Nn + 255) / 256), dim3(256), 0, stream>>>(NN1, NN0, NNbf, Nn * Nn);

  // --- av = attn @ v ; gated; out = (av*gate)@Wout^T + bout ; s = out*x ---
  cvt(h_f, h_bf, Nn * H2);
  GEMM(false, 0, NNbf, Nn, h_bf, H2, av_f, DSUP, (const float*)nullptr, 1.0f, Nn, DSUP, Nn);
  k_avgate<<<dim3((Nn * DSUP + 255) / 256), dim3(256), 0, stream>>>(av_f, h_f, avg_bf, Nn * DSUP);
  GEMM(true, 0, avg_bf, DSUP, Wout_bf, DSUP, o_f, DSUP, gbout, 1.0f, Nn, DSUP, DSUP);
  k_mul_bf16<<<dim3((Nn * DSUP + 255) / 256), dim3(256), 0, stream>>>(o_f, sup, s_bf, Nn * DSUP);

  // --- K, V, Q ---
  GEMM(true, 0, s_bf, DSUP, Wk_bf, DSUP, K_f, DIN, bk, 1.0f, Nn, DIN, DSUP);
  GEMM(true, 0, s_bf, DSUP, Wv_bf, DSUP, V_f, DIN, bv, 1.0f, Nn, DIN, DSUP);
  GEMM(true, 0, node_bf, DIN, Wq_bf, DIN, Q_f, DIN, bq, 1.0f, Nn, DIN, DIN);
  cvt(K_f, K_bf, Nn * DIN);
  cvt(V_f, V_bf, Nn * DIN);
  cvt(Q_f, Q_bf, Nn * DIN);

  // --- B1 = K * (Q + relu(softmax(K@K^T)@Q)) * Q ---
  GEMM(true, 0, K_bf, DIN, K_bf, DIN, NN0, Nn, (const float*)nullptr, 1.0f, Nn, Nn, DIN);
  row_softmax_4096<__bf16><<<Nn, 256, 0, stream>>>(NN0, NNbf);
  GEMM(false, 0, NNbf, Nn, Q_bf, DIN, G_f, DIN, (const float*)nullptr, 1.0f, Nn, DIN, Nn);
  k_bcombine<<<dim3((Nn * DIN + 255) / 256), dim3(256), 0, stream>>>(
      K_f, Q_f, G_f, B1_f, B1_bf, Nn * DIN);

  // --- B2 = V * (Q + relu(softmax(V@V^T)@Q)) * Q ---
  GEMM(true, 0, V_bf, DIN, V_bf, DIN, NN0, Nn, (const float*)nullptr, 1.0f, Nn, Nn, DIN);
  row_softmax_4096<__bf16><<<Nn, 256, 0, stream>>>(NN0, NNbf);
  GEMM(false, 0, NNbf, Nn, Q_bf, DIN, G_f, DIN, (const float*)nullptr, 1.0f, Nn, DIN, Nn);
  k_bcombine<<<dim3((Nn * DIN + 255) / 256), dim3(256), 0, stream>>>(
      V_f, Q_f, G_f, B2_f, B2_bf, Nn * DIN);

  // --- gt = sigmoid(B1@Wg^T + bg) ; Gs = softmax((gt*K)@((1-gt)*Q)^T / 16) ---
  GEMM(true, 2, B1_bf, DIN, Wg_bf, DIN, gt_f, DIN, bg, 1.0f, Nn, DIN, DIN);
  k_gtsplit<<<dim3((Nn * DIN + 255) / 256), dim3(256), 0, stream>>>(
      gt_f, K_f, Q_f, X1_bf, X2_bf, Nn * DIN);
  GEMM(true, 0, X1_bf, DIN, X2_bf, DIN, NN0, Nn, (const float*)nullptr, 1.0f / 16.0f, Nn, Nn, DIN);
  row_softmax_4096<__bf16><<<Nn, 256, 0, stream>>>(NN0, NNbf);

  // --- out = normalize(sigmoid(B1*B1*(Gs@B2))) ---
  GEMM(false, 0, NNbf, Nn, B2_bf, DIN, GB2_f, DIN, (const float*)nullptr, 1.0f, Nn, DIN, Nn);
  k_finalize<<<dim3(Nn), dim3(256), 0, stream>>>(B1_f, GB2_f, (float*)d_out);
}
